// TripletLoss_42949673749
// MI455X (gfx1250) — matrix operand order repaired and checked by reading, not compiled, not run on hardware
//
#include <hip/hip_runtime.h>
#include <hip/hip_bf16.h>

// CDNA5 (gfx1250) fused batch-hard triplet loss.
//   Phase 1: L2-normalize rows of E (f32) -> f16 matrix in workspace (2 MB, L2-resident).
//   Phase 2: tiled Gram matrix via v_wmma_f32_16x16x32_f16 with fused hard-mining epilogue.
//            32 rows per block (two 16-row A tiles); each B fragment feeds two WMMAs
//            (two independent accumulator chains). Mining runs on integer bit-pattern
//            min/max (no canonicalize overhead). The column-tile loop counter is forced
//            scalar (readfirstlane) so the diagonal-exclusion peel is a true s_cbranch.
//   Phase 3: deterministic reduction of per-block partials -> scalar loss.

typedef __attribute__((ext_vector_type(16))) _Float16 v16h;
typedef __attribute__((ext_vector_type(4)))  _Float16 v4h;
typedef __attribute__((ext_vector_type(8)))  float    v8f;

#define TL_MARGIN 0.5f
#define TL_EPS 1e-12f
#define POS_SENTINEL ((int)0x80000000)   // INT_MIN: "no positive seen"
#define NEG_SENTINEL ((int)0x7FFFFFFF)   // INT_MAX: "no negative seen"

union F16x16 {
    v16h v;
    struct { uint4 lo; uint4 hi; } u;
};

// ---------------- Phase 1: normalize rows, emit f16 ----------------
// One wave per row (D = 128 -> 4 floats per lane).
__global__ __launch_bounds__(256)
void tl_normalize_f16(const float* __restrict__ emb, _Float16* __restrict__ eh) {
    const int row  = blockIdx.x * 8 + (threadIdx.x >> 5);
    const int lane = threadIdx.x & 31;
    const float4 x = *(const float4*)(emb + (size_t)row * 128 + lane * 4);
    float ss = x.x * x.x + x.y * x.y + x.z * x.z + x.w * x.w;
    #pragma unroll
    for (int m = 16; m >= 1; m >>= 1) ss += __shfl_xor(ss, m, 32);
    const float inv = 1.0f / fmaxf(sqrtf(ss), TL_EPS);
    v4h o;
    o[0] = (_Float16)(x.x * inv);
    o[1] = (_Float16)(x.y * inv);
    o[2] = (_Float16)(x.z * inv);
    o[3] = (_Float16)(x.w * inv);
    *(v4h*)(eh + (size_t)row * 128 + lane * 4) = o;
}

// Mining epilogue for one 16x16 C tile (8 elements per lane).
// s = 2 - 2g clamped to [0,8] via one v_med3_f32, then integer min/max on the
// bit pattern (monotone for non-negative floats, no canonicalization needed).
template <bool DIAG>
__device__ __forceinline__
void tl_update(const v8f& c, const int* rowLab, int iBase, int j, int colLab,
               int* posB, int* negB) {
    #pragma unroll
    for (int r = 0; r < 8; ++r) {
        const float s    = 2.0f - 2.0f * c[r];
        const float sc   = __builtin_amdgcn_fmed3f(s, 0.0f, 8.0f);
        const int   bits = __float_as_int(sc);
        const bool  same = (rowLab[r] == colLab);
        bool posOk = same;
        if (DIAG) posOk = same && ((iBase + r) != j);   // exclude self-pair
        const int pc = posOk ? bits : POS_SENTINEL;
        const int nc = same  ? NEG_SENTINEL : bits;
        posB[r] = max(posB[r], pc);
        negB[r] = min(negB[r], nc);
    }
}

// ---------------- Phase 2: WMMA Gram tiles + fused hard mining ----------------
// Block = 8 wave32, owns a 32-row strip (two 16-row A tiles). Wave w handles column
// tiles w, w+8, ... Each B fragment is consumed by two WMMAs (one per row tile).
// A fragment (16x32 f16): lane<16 -> row rowBase+lane, K = {8h..8h+7, 16+8h..16+8h+7}
// B fragment (32x16 f16): lane    -> col colBase+(lane&15), K = {8h..8h+7, 16+8h..16+8h+7}
__global__ __launch_bounds__(256)
void tl_mine(const _Float16* __restrict__ eh,
             const int* __restrict__ labels,
             float* __restrict__ blockLoss,
             float* __restrict__ blockValid,
             int N) {
    __shared__ int posS[8][32];
    __shared__ int negS[8][32];

    const int tid  = threadIdx.x;
    // Scalar (SGPR) wave index: makes the col-tile loop counter provably uniform so the
    // diagonal peel compiles to s_cmp + s_cbranch instead of divergent exec masking.
    const int waveS = __builtin_amdgcn_readfirstlane(tid >> 5);
    const int lane = tid & 31;
    const int l16  = lane & 15;
    const int h    = lane >> 4;           // half-wave select
    const int rowBase = blockIdx.x * 32;

    // Preload A fragments for both row tiles, all 4 K-chunks (reused for every col tile).
    F16x16 A[2][4];
    #pragma unroll
    for (int rt = 0; rt < 2; ++rt) {
        const _Float16* arow = eh + (size_t)(rowBase + rt * 16 + l16) * 128;
        #pragma unroll
        for (int q = 0; q < 4; ++q) {
            const int kc = q * 32;
            A[rt][q].u.lo = *(const uint4*)(arow + kc + 8 * h);
            A[rt][q].u.hi = *(const uint4*)(arow + kc + 16 + 8 * h);
        }
    }

    // Row labels for the 8 C-matrix rows per tile this lane sees (M = r + 8*h).
    int rowLab[2][8];
    #pragma unroll
    for (int rt = 0; rt < 2; ++rt)
        #pragma unroll
        for (int r = 0; r < 8; ++r)
            rowLab[rt][r] = labels[rowBase + rt * 16 + r + 8 * h];

    int posB[2][8], negB[2][8];
    #pragma unroll
    for (int rt = 0; rt < 2; ++rt)
        #pragma unroll
        for (int r = 0; r < 8; ++r) { posB[rt][r] = POS_SENTINEL; negB[rt][r] = NEG_SENTINEL; }

    const int dct0 = rowBase >> 4;        // first of the 2 col tiles that touch the diagonal
    const int nColTiles = N >> 4;
    for (int ct = waveS; ct < nColTiles; ct += 8) {
        const int colBase = ct * 16;      // scalar
        const int j = colBase + l16;
        const int colLab = labels[j];

        // B fragment: two 8-half runs per lane per K-chunk (K = 8h.., 16+8h..).
        const _Float16* brow = eh + (size_t)j * 128 + 8 * h;
        F16x16 B[4];
        #pragma unroll
        for (int q = 0; q < 4; ++q) {
            B[q].u.lo = *(const uint4*)(brow + q * 32);
            B[q].u.hi = *(const uint4*)(brow + q * 32 + 16);
        }

        // Two independent accumulator chains, interleaved across K-chunks.
        v8f c0 = {}, c1 = {};
        #pragma unroll
        for (int q = 0; q < 4; ++q) {
            c0 = __builtin_amdgcn_wmma_f32_16x16x32_f16(
                     false, A[0][q].v, false, B[q].v, (short)0, c0, false, false);
            c1 = __builtin_amdgcn_wmma_f32_16x16x32_f16(
                     false, A[1][q].v, false, B[q].v, (short)0, c1, false, false);
        }

        // Scalar branch: only 2 of the 512 column tiles can contain diagonal elements.
        if (ct == dct0 || ct == dct0 + 1) {
            tl_update<true >(c0, rowLab[0], rowBase +      8 * h, j, colLab, posB[0], negB[0]);
            tl_update<true >(c1, rowLab[1], rowBase + 16 + 8 * h, j, colLab, posB[1], negB[1]);
        } else {
            tl_update<false>(c0, rowLab[0], 0, 0, colLab, posB[0], negB[0]);
            tl_update<false>(c1, rowLab[1], 0, 0, colLab, posB[1], negB[1]);
        }
    }

    // Reduce across the 16 lanes of each half-wave (N dimension of the tile).
    #pragma unroll
    for (int rt = 0; rt < 2; ++rt)
        #pragma unroll
        for (int r = 0; r < 8; ++r) {
            #pragma unroll
            for (int m = 1; m < 16; m <<= 1) {
                posB[rt][r] = max(posB[rt][r], __shfl_xor(posB[rt][r], m, 32));
                negB[rt][r] = min(negB[rt][r], __shfl_xor(negB[rt][r], m, 32));
            }
        }
    if (l16 == 0) {
        #pragma unroll
        for (int rt = 0; rt < 2; ++rt)
            #pragma unroll
            for (int r = 0; r < 8; ++r) {
                posS[waveS][rt * 16 + r + 8 * h] = posB[rt][r];
                negS[waveS][rt * 16 + r + 8 * h] = negB[rt][r];
            }
    }
    __syncthreads();

    // Combine the 8 waves' column ranges; finish per-row loss; one partial per block.
    if (tid < 32) {
        int pb = POS_SENTINEL, nb = NEG_SENTINEL;
        #pragma unroll
        for (int w = 0; w < 8; ++w) {
            pb = max(pb, posS[w][tid]);
            nb = min(nb, negS[w][tid]);
        }
        const bool valid = (pb != POS_SENTINEL) && (nb != NEG_SENTINEL);
        const float sp = __int_as_float(pb);   // clamped to [0,8]; -0.0f if invalid
        const float sn = __int_as_float(nb);
        const float pd = sqrtf(fmaxf(sp, TL_EPS));
        const float nd = sqrtf(fmaxf(sn, TL_EPS));
        float loss = valid ? fmaxf(pd - nd + TL_MARGIN, 0.0f) : 0.0f;
        float cnt  = valid ? 1.0f : 0.0f;
        #pragma unroll
        for (int m = 1; m < 32; m <<= 1) {
            loss += __shfl_xor(loss, m, 32);
            cnt  += __shfl_xor(cnt,  m, 32);
        }
        if (tid == 0) {
            blockLoss[blockIdx.x]  = loss;
            blockValid[blockIdx.x] = cnt;
        }
    }
}

// ---------------- Phase 3: deterministic finalize ----------------
__global__ __launch_bounds__(32)
void tl_finalize(const float* __restrict__ blockLoss,
                 const float* __restrict__ blockValid,
                 float* __restrict__ out, int nBlocks) {
    const int lane = threadIdx.x;
    float s = 0.0f, c = 0.0f;
    for (int i = lane; i < nBlocks; i += 32) { s += blockLoss[i]; c += blockValid[i]; }
    #pragma unroll
    for (int m = 16; m >= 1; m >>= 1) { s += __shfl_xor(s, m, 32); c += __shfl_xor(c, m, 32); }
    if (lane == 0) out[0] = (c > 0.0f) ? (s / fmaxf(c, 1.0f)) : 0.0f;
}

extern "C" void kernel_launch(void* const* d_in, const int* in_sizes, int n_in,
                              void* d_out, int out_size, void* d_ws, size_t ws_size,
                              hipStream_t stream) {
    const float* emb    = (const float*)d_in[0];
    const int*   labels = (const int*)d_in[1];
    float*       out    = (float*)d_out;

    const int N = in_sizes[1];          // 8192 (D fixed at 128 per reference)
    const int nBlocks = N / 32;         // 256 strips of 32 rows

    char*  ws         = (char*)d_ws;
    float* blockLoss  = (float*)ws;
    float* blockValid = blockLoss + nBlocks;
    const size_t ehOff = ((size_t)(2 * nBlocks * sizeof(float)) + 4095) & ~(size_t)4095;
    _Float16* eh = (_Float16*)(ws + ehOff);   // N*128 halves = 2 MB, L2-resident

    tl_normalize_f16<<<N / 8, 256, 0, stream>>>(emb, eh);
    tl_mine<<<nBlocks, 256, 0, stream>>>(eh, labels, blockLoss, blockValid, N);
    tl_finalize<<<1, 32, 0, stream>>>(blockLoss, blockValid, out, nBlocks);
}
